// SelfAttention_65481071407513
// MI455X (gfx1250) — matrix operand order repaired and checked by reading, not compiled, not run on hardware
//
#include <hip/hip_runtime.h>
#include <hip/hip_bf16.h>
#include <math.h>

#define DIM 1024
#define SEQ 2048
#define NB  4
#define PROBS_LD 2056  // 2048 + 8 halfs pad; row pitch 4112B (16B aligned)

typedef _Float16 half_t;
typedef __attribute__((ext_vector_type(16))) _Float16 v16h;
typedef __attribute__((ext_vector_type(8)))  _Float16 v8h;
typedef __attribute__((ext_vector_type(8)))  float    v8f;

static __device__ __forceinline__ int lane_id() { return threadIdx.x & 31; }

// A-fragment (16x32 f16, MxK), source row-major with leading dim ld.
// lanes 0-15: row M=lane, halves0-15 = K 0..15;
// lanes 16-31: same rows, K 16..31.
static __device__ __forceinline__ v16h load_a_frag(const half_t* tile, int ld) {
  const int lane = lane_id();
  const int row  = lane & 15;
  const int kb   = (lane & 16) ? 16 : 0;
  const v8h* p = reinterpret_cast<const v8h*>(tile + row * ld + kb);
  v8h lo = p[0];     // K = kb .. kb+7
  v8h hi = p[1];     // K = kb+8 .. kb+15
  return __builtin_shufflevector(lo, hi, 0,1,2,3,4,5,6,7,8,9,10,11,12,13,14,15);
}

// B-fragment (32x16 f16, KxN), source stored K-major per column:
// B[k][n] = tileT[n*ld + k].  lanes 0-15: col=lane, K 0..15; lanes 16-31: K 16..31.
static __device__ __forceinline__ v16h load_b_frag(const half_t* tileT, int ld) {
  const int lane = lane_id();
  const int col  = lane & 15;
  const int kb   = (lane & 16) ? 16 : 0;
  return *reinterpret_cast<const v16h*>(tileT + col * ld + kb);
}

static __device__ __forceinline__ void wmma_f16(v16h a, v16h b, v8f& c) {
  c = __builtin_amdgcn_wmma_f32_16x16x32_f16(false, a, false, b, (short)0, c,
                                             false, false);
}

// ---------------- prep: fp32 -> f16 conversions / transposes ----------------

__global__ void prep_weights(const float* __restrict__ rot,
                             const float* __restrict__ ent,
                             half_t* __restrict__ rotT,
                             half_t* __restrict__ entT) {
  __shared__ float tile[16][17];
  const float* src = blockIdx.z ? ent : rot;
  half_t*      dst = blockIdx.z ? entT : rotT;
  const int tx = threadIdx.x, ty = threadIdx.y;
  const int d0 = blockIdx.y * 16, e0 = blockIdx.x * 16;
  tile[ty][tx] = src[(size_t)(d0 + ty) * DIM + e0 + tx];
  __syncthreads();
  dst[(size_t)(e0 + ty) * DIM + d0 + tx] = (half_t)tile[tx][ty];
}

__global__ void prep_x(const float* __restrict__ x,
                       half_t* __restrict__ x_h,
                       half_t* __restrict__ x_t) {
  __shared__ float tile[16][17];
  const int tx = threadIdx.x, ty = threadIdx.y;
  const int b  = blockIdx.z;
  const int s0 = blockIdx.y * 16, d0 = blockIdx.x * 16;
  const float v = x[((size_t)b * SEQ + s0 + ty) * DIM + d0 + tx];
  x_h[((size_t)b * SEQ + s0 + ty) * DIM + d0 + tx] = (half_t)v;
  tile[ty][tx] = v;
  __syncthreads();
  x_t[((size_t)b * DIM + d0 + ty) * SEQ + s0 + tx] = (half_t)tile[tx][ty];
}

// ---------------- projections: q = x*R * (1/sqrt(D)),  k = x*E ----------------
// grid (64, 8, 2), block 256 (8 waves); wave tile 32x64 (2 M-subtiles share the
// 4 B-fragments), WG tile 128x128.  K loop unrolled 2x with even/odd fragment
// buffers (no register rotation copies): loads for the next 32-step are always
// in flight behind the current WMMA burst.

__global__ __launch_bounds__(256) void proj_gemm(
    const half_t* __restrict__ x_h, const half_t* __restrict__ rotT,
    const half_t* __restrict__ entT, half_t* __restrict__ q_h,
    half_t* __restrict__ k_h) {
  const int w    = threadIdx.x >> 5;
  const int lane = threadIdx.x & 31;
  const half_t* wT  = blockIdx.z ? entT : rotT;
  half_t*       dst = blockIdx.z ? k_h  : q_h;
  const float oscale = blockIdx.z ? 1.0f : 0.03125f;  // 1/sqrt(1024) into q
  const int m0 = blockIdx.x * 128 + (w & 3) * 32;
  const int n0 = blockIdx.y * 128 + (w >> 2) * 64;

  const half_t* abase = x_h + (size_t)m0 * DIM;
  const half_t* bbase = wT + (size_t)n0 * DIM;

  v8f acc[8] = {};
  v16h ea0, ea1, eb[4];  // even buffer (kk)
  v16h oa0, oa1, ob[4];  // odd buffer  (kk+32)

  ea0 = load_a_frag(abase, DIM);
  ea1 = load_a_frag(abase + 16 * DIM, DIM);
#pragma unroll
  for (int t = 0; t < 4; ++t) eb[t] = load_b_frag(bbase + (size_t)(t * 16) * DIM, DIM);

#pragma unroll 1
  for (int kk = 0; kk < DIM; kk += 64) {
    // prefetch odd half (kk+32) -- always in range (DIM % 64 == 0)
    oa0 = load_a_frag(abase + kk + 32, DIM);
    oa1 = load_a_frag(abase + 16 * DIM + kk + 32, DIM);
#pragma unroll
    for (int t = 0; t < 4; ++t)
      ob[t] = load_b_frag(bbase + (size_t)(t * 16) * DIM + kk + 32, DIM);
    // compute even half
#pragma unroll
    for (int t = 0; t < 4; ++t) wmma_f16(ea0, eb[t], acc[t]);
#pragma unroll
    for (int t = 0; t < 4; ++t) wmma_f16(ea1, eb[t], acc[4 + t]);
    // prefetch next even half (kk+64)
    if (kk + 64 < DIM) {
      ea0 = load_a_frag(abase + kk + 64, DIM);
      ea1 = load_a_frag(abase + 16 * DIM + kk + 64, DIM);
#pragma unroll
      for (int t = 0; t < 4; ++t)
        eb[t] = load_b_frag(bbase + (size_t)(t * 16) * DIM + kk + 64, DIM);
    }
    // compute odd half
#pragma unroll
    for (int t = 0; t < 4; ++t) wmma_f16(oa0, ob[t], acc[t]);
#pragma unroll
    for (int t = 0; t < 4; ++t) wmma_f16(oa1, ob[t], acc[4 + t]);
  }

  const int col   = lane & 15;
  const int rbase = (lane & 16) ? 8 : 0;
#pragma unroll
  for (int mi = 0; mi < 2; ++mi)
#pragma unroll
    for (int t = 0; t < 4; ++t)
#pragma unroll
      for (int r = 0; r < 8; ++r)
        dst[(size_t)(m0 + mi * 16 + rbase + r) * DIM + n0 + t * 16 + col] =
            (half_t)(acc[mi * 4 + t][r] * oscale);
}

// ---------------- fused attention: scores -> softmax -> attn*x ----------------
// grid (SEQ/16, NB), block 512 (16 waves), dynamic LDS = 16*PROBS_LD halfs.
// Phase 1: wave w computes score tile rows[16] x keys[w*128 .. w*128+128).
// Phase 2: cross-wave softmax (shuffle + 1KB LDS reduction).
// Phase 3: probs f16 -> LDS.  Phase 4: wave w computes out columns [w*64, w*64+64).

__global__ __launch_bounds__(512) void attention(
    const half_t* __restrict__ q_h, const half_t* __restrict__ k_h,
    const half_t* __restrict__ x_t, float* __restrict__ out) {
  extern __shared__ half_t probs[];
  __shared__ float red[256];

  const int w     = threadIdx.x >> 5;
  const int lane  = threadIdx.x & 31;
  const int qb    = blockIdx.x;
  const int b     = blockIdx.y;
  const int col   = lane & 15;
  const int rbase = (lane & 16) ? 8 : 0;

  const half_t* qbase = q_h + ((size_t)b * SEQ + qb * 16) * DIM;
  const half_t* kbase = k_h + ((size_t)b * SEQ + w * 128) * DIM;

  // ---- Phase 1: scores (already scaled; q carries 1/sqrt(D)) ----
  v8f sacc[8] = {};
#pragma unroll 1
  for (int kk = 0; kk < DIM; kk += 32) {
    v16h a = load_a_frag(qbase + kk, DIM);
    v16h bb[8];
#pragma unroll
    for (int kt = 0; kt < 8; ++kt)
      bb[kt] = load_b_frag(kbase + (size_t)(kt * 16) * DIM + kk, DIM);
#pragma unroll
    for (int kt = 0; kt < 8; ++kt) wmma_f16(a, bb[kt], sacc[kt]);
  }

  // ---- Phase 2: softmax stats ----
  float lm[8];
#pragma unroll
  for (int r = 0; r < 8; ++r) {
    float m = sacc[0][r];
#pragma unroll
    for (int kt = 1; kt < 8; ++kt) m = fmaxf(m, sacc[kt][r]);
    lm[r] = m;
  }
#pragma unroll
  for (int off = 8; off >= 1; off >>= 1)
#pragma unroll
    for (int r = 0; r < 8; ++r)
      lm[r] = fmaxf(lm[r], __shfl_xor(lm[r], off, 32));
  if (lane == 0 || lane == 16) {
#pragma unroll
    for (int r = 0; r < 8; ++r) red[w * 16 + rbase + r] = lm[r];
  }
  __syncthreads();
  float rm[8];
#pragma unroll
  for (int r = 0; r < 8; ++r) {
    float m = red[rbase + r];
    for (int w2 = 1; w2 < 16; ++w2) m = fmaxf(m, red[w2 * 16 + rbase + r]);
    rm[r] = m;
  }
  __syncthreads();  // all reads of red done before sum phase reuses it

  float ls[8] = {};
#pragma unroll
  for (int kt = 0; kt < 8; ++kt)
#pragma unroll
    for (int r = 0; r < 8; ++r) {
      float e = __expf(sacc[kt][r] - rm[r]);
      sacc[kt][r] = e;
      ls[r] += e;
    }
#pragma unroll
  for (int off = 8; off >= 1; off >>= 1)
#pragma unroll
    for (int r = 0; r < 8; ++r)
      ls[r] += __shfl_xor(ls[r], off, 32);
  if (lane == 0 || lane == 16) {
#pragma unroll
    for (int r = 0; r < 8; ++r) red[w * 16 + rbase + r] = ls[r];
  }
  __syncthreads();
  float inv[8];
#pragma unroll
  for (int r = 0; r < 8; ++r) {
    float s = 0.f;
    for (int w2 = 0; w2 < 16; ++w2) s += red[w2 * 16 + rbase + r];
    inv[r] = 1.0f / s;
  }

  // ---- Phase 3: probs -> LDS (f16) ----
#pragma unroll
  for (int kt = 0; kt < 8; ++kt)
#pragma unroll
    for (int r = 0; r < 8; ++r)
      probs[(size_t)(rbase + r) * PROBS_LD + w * 128 + kt * 16 + col] =
          (half_t)(sacc[kt][r] * inv[r]);
  __syncthreads();

  // ---- Phase 4: out = probs @ x  (x_t gives contiguous B fragments) ----
  v8f oacc[4] = {};
  const half_t* vbase = x_t + ((size_t)b * DIM + w * 64) * SEQ;
#pragma unroll 1
  for (int kk = 0; kk < SEQ; kk += 32) {
    v16h a = load_a_frag(probs + kk, PROBS_LD);
    v16h bb[4];
#pragma unroll
    for (int t = 0; t < 4; ++t)
      bb[t] = load_b_frag(vbase + (size_t)(t * 16) * SEQ + kk, SEQ);
#pragma unroll
    for (int t = 0; t < 4; ++t) wmma_f16(a, bb[t], oacc[t]);
  }
  float* obase = out + ((size_t)b * SEQ + qb * 16) * DIM;
#pragma unroll
  for (int t = 0; t < 4; ++t)
#pragma unroll
    for (int r = 0; r < 8; ++r)
      obase[(size_t)(rbase + r) * DIM + w * 64 + t * 16 + col] = oacc[t][r];
}

// ---------------- host ----------------

extern "C" void kernel_launch(void* const* d_in, const int* in_sizes, int n_in,
                              void* d_out, int out_size, void* d_ws, size_t ws_size,
                              hipStream_t stream) {
  (void)in_sizes; (void)n_in; (void)out_size; (void)ws_size;
  const float* x   = (const float*)d_in[0];
  const float* rot = (const float*)d_in[1];
  const float* ent = (const float*)d_in[2];
  float* out = (float*)d_out;

  char* ws = (char*)d_ws;
  // f16 workspace layout (bytes):
  half_t* x_h  = (half_t*)(ws + 0);          // [8192][1024]        16 MB
  half_t* x_t  = (half_t*)(ws + 16777216);   // [4][1024][2048]     16 MB
  half_t* rotT = (half_t*)(ws + 33554432);   // [1024][1024]         2 MB
  half_t* entT = (half_t*)(ws + 35651584);   // [1024][1024]         2 MB
  half_t* q_h  = (half_t*)(ws + 37748736);   // [8192][1024]        16 MB
  half_t* k_h  = (half_t*)(ws + 54525952);   // [8192][1024]        16 MB  (end 68 MB)

  prep_weights<<<dim3(64, 64, 2), dim3(16, 16), 0, stream>>>(rot, ent, rotT, entT);
  prep_x<<<dim3(64, 128, 4), dim3(16, 16), 0, stream>>>(x, x_h, x_t);
  proj_gemm<<<dim3(64, 8, 2), 256, 0, stream>>>(x_h, rotT, entT, q_h, k_h);
  attention<<<dim3(SEQ / 16, NB), 512, 16 * PROBS_LD * sizeof(half_t), stream>>>(
      q_h, k_h, x_t, out);
}
